// RaggedConvolutionTranspose_45612552683660
// MI455X (gfx1250) — matrix operand order repaired and checked by reading, not compiled, not run on hardware
//
#include <hip/hip_runtime.h>

typedef __attribute__((ext_vector_type(16))) _Float16 v16h;
typedef __attribute__((ext_vector_type(8)))  float    v8f;

#define NI_N   50000
#define NO_N   50000
#define FI_K   64
#define UNITS_ 64
#define D_     3
#define NH     (UNITS_ * D_)   // 192

// ---------------------------------------------------------------------------
// Kernel 1: H = X @ W + b   using v_wmma_f32_16x16x32_f16
//   X: [NI,64] f32 (converted to f16 on load), W: [64,192] f32, H: [NI,192] f32
//   grid.x = NI/16 M-tiles; block = 128 threads (4 waves).
//   Wave w computes N-tiles {w, w+4, w+8}; A fragment loaded once per wave.
// ---------------------------------------------------------------------------
__global__ void __launch_bounds__(128)
ragged_dense_proj_wmma(const float* __restrict__ X,
                       const float* __restrict__ W,
                       const float* __restrict__ bias,
                       float* __restrict__ H)
{
    const int lane  = threadIdx.x & 31;
    const int wave  = threadIdx.x >> 5;
    const int mtile = blockIdx.x;
    const int m     = lane & 15;        // M within tile (A) / N within tile (B,C,D)
    const int grp   = lane >> 4;        // lane group 0/1
    const int arow  = mtile * 16 + m;

    // A fragments for K-steps 0..31 and 32..63 (ISA 16-bit A layout, 16x32):
    //   VGPR v (halves j=2v,2v+1): Kbase = (v<4 ? 2v : 16+2(v-4)) + 8*grp
    v16h afrag[2];
#pragma unroll
    for (int kb = 0; kb < 2; ++kb) {
#pragma unroll
        for (int j = 0; j < 16; ++j) {
            const int v     = j >> 1;
            const int kbase = (v < 4 ? 2 * v : 16 + 2 * (v - 4)) + 8 * grp;
            const int k     = kb * 32 + kbase + (j & 1);
            afrag[kb][j] = (_Float16)X[(size_t)arow * FI_K + k];
        }
    }

#pragma unroll
    for (int t = 0; t < 3; ++t) {
        const int ntile = wave + 4 * t;        // 0..11  (192/16 N-tiles)
        const int n     = ntile * 16 + m;

        // C preloaded with bias (constant along M -> every C VGPR = b[n])
        v8f c;
        const float bv = bias[n];
#pragma unroll
        for (int r = 0; r < 8; ++r) c[r] = bv;

#pragma unroll
        for (int kb = 0; kb < 2; ++kb) {
            // B fragment (32x16, 16-bit): lane n-col; halves j -> K = (j<8 ? 8*grp + j : 16 + 8*grp + (j-8))
            v16h bfrag;
#pragma unroll
            for (int j = 0; j < 16; ++j) {
                const int k = kb * 32 + (j < 8 ? 8 * grp + j : 16 + 8 * grp + (j - 8));
                bfrag[j] = (_Float16)W[(size_t)k * NH + n];
            }
            c = __builtin_amdgcn_wmma_f32_16x16x32_f16(
                    /*neg_a=*/false, afrag[kb],
                    /*neg_b=*/false, bfrag,
                    /*c_mod=*/(short)0, c,
                    /*reuse_a=*/false, /*reuse_b=*/false);
        }

        // D layout: VGPR r -> M = r + 8*grp, N = lane%16
#pragma unroll
        for (int r = 0; r < 8; ++r) {
            const int orow = mtile * 16 + r + 8 * grp;
            H[(size_t)orow * NH + n] = c[r];
        }
    }
}

// ---------------------------------------------------------------------------
// Kernel 2: per-edge gather of h rows (L2-resident), contract over coord dim,
//           segment-sum over each node's edge range from row_splits.
//   One wave32 per output node; lane L owns units 2L,2L+1 -> its 6 h values
//   are contiguous, so the wave reads 192 contiguous floats/edge (coalesced).
// ---------------------------------------------------------------------------
__global__ void __launch_bounds__(256)
ragged_edge_contract(const float* __restrict__ H,
                     const float* __restrict__ coord,   // [E,3]
                     const int*   __restrict__ idx,     // [E]
                     const int*   __restrict__ rs,      // [NO+1]
                     float*       __restrict__ out)     // [NO,64]
{
    const int lane = threadIdx.x & 31;
    const int wave = threadIdx.x >> 5;
    const int o    = blockIdx.x * 8 + wave;
    if (o >= NO_N) return;

    const int e0 = rs[o];
    const int e1 = rs[o + 1];

    float acc0 = 0.0f, acc1 = 0.0f;
    for (int e = e0; e < e1; ++e) {
        const int src = idx[e];
        // 6 contiguous floats for this lane: h[src, 6L .. 6L+5] (8B aligned)
        const float2* gp = (const float2*)(H + (size_t)src * NH) + 3 * lane;
        const float2 g01 = gp[0];
        const float2 g23 = gp[1];
        const float2 g45 = gp[2];
        const float c0 = coord[3 * (size_t)e + 0];
        const float c1 = coord[3 * (size_t)e + 1];
        const float c2 = coord[3 * (size_t)e + 2];
        acc0 += g01.x * c0 + g01.y * c1 + g23.x * c2;   // unit 2L
        acc1 += g23.y * c0 + g45.x * c1 + g45.y * c2;   // unit 2L+1
    }
    out[(size_t)o * UNITS_ + 2 * lane]     = acc0;
    out[(size_t)o * UNITS_ + 2 * lane + 1] = acc1;
}

// ---------------------------------------------------------------------------
extern "C" void kernel_launch(void* const* d_in, const int* in_sizes, int n_in,
                              void* d_out, int out_size, void* d_ws, size_t ws_size,
                              hipStream_t stream)
{
    (void)in_sizes; (void)n_in; (void)out_size; (void)ws_size;

    const float* X     = (const float*)d_in[0];  // node_features [NI,64]
    const float* coord = (const float*)d_in[1];  // coord_features [E,3]
    const int*   idx   = (const int*)  d_in[2];  // indices [E]
    const int*   rs    = (const int*)  d_in[3];  // row_splits [NO+1]
    const float* W     = (const float*)d_in[4];  // [64,192]
    const float* b     = (const float*)d_in[5];  // [192]
    float* out = (float*)d_out;                  // [NO,64]
    float* H   = (float*)d_ws;                   // scratch: [NI,192] f32 = 38.4 MB

    // Phase 1: WMMA projection (h stays hot in 192 MB L2 for phase 2 gathers)
    ragged_dense_proj_wmma<<<NI_N / 16, 128, 0, stream>>>(X, W, b, H);

    // Phase 2: gather + contract + ragged segment-sum
    ragged_edge_contract<<<(NO_N + 7) / 8, 256, 0, stream>>>(H, coord, idx, rs, out);
}
